// topk_mil_53661321396717
// MI455X (gfx1250) — matrix operand order repaired and checked by reading, not compiled, not run on hardware
//
#include <hip/hip_runtime.h>
#include <hip/hip_bf16.h>
#include <math.h>

// ---------------------------------------------------------------------------
// Problem constants (match reference)
// ---------------------------------------------------------------------------
#define B_BATCH 8
#define N_PATCH 16384
#define F_DIM   1024
#define Z_DIM   256
#define TOPK    20
#define BN_EPS  1e-5f

#define ROWS_TOTAL      (B_BATCH * N_PATCH)   // 131072
#define ROWS_PER_BLOCK  128                    // block covers 128 rows x 256 cols
#define KSTEP           32                     // K per WMMA
#define NTILES          (Z_DIM / 16)           // 16 N-tiles of 16
#define KTILES          (F_DIM / KSTEP)        // 32 K-steps
// Packed bf16 W_enc fragments: (KTILES*NTILES) tiles * 32 lanes * 32 bytes
#define WENC_FRAG_BYTES (KTILES * NTILES * 32 * 32)   // 524288 = 512 KB
#define B_KSTEP_BYTES   (NTILES * 32 * 32)            // 16 KB per k-step
#define A_KSTEP_BYTES   (8 * 32 * 32)                 // 8 M-tiles -> 8 KB per k-step

typedef __attribute__((ext_vector_type(16))) __bf16 bf16x16;
typedef __attribute__((ext_vector_type(8)))  float  f32x8;
typedef __attribute__((ext_vector_type(4)))  int    i32x4;

// Builtin wants v4i pointers: AS1 (global) source, AS3 (LDS) destination.
typedef __attribute__((address_space(1))) i32x4 as1_i32x4;
typedef __attribute__((address_space(3))) i32x4 as3_i32x4;

// ---------------------------------------------------------------------------
// CDNA5 async global->LDS copy (ASYNCcnt path), with safe fallback.
// ---------------------------------------------------------------------------
__device__ __forceinline__ void async_copy_b128(const void* g, void* l) {
#if __has_builtin(__builtin_amdgcn_global_load_async_to_lds_b128)
    __builtin_amdgcn_global_load_async_to_lds_b128(
        (as1_i32x4*)g, (as3_i32x4*)l, 0, 0);
#else
    *(float4*)l = *(const float4*)g;
#endif
}

__device__ __forceinline__ void wait_async_zero() {
    asm volatile("s_wait_asynccnt 0x0" ::: "memory");
}

// ---------------------------------------------------------------------------
// Kernel 1: repack W_enc (f32 [F,Z] row-major) into bf16 B-matrix fragments.
// B fragment (32x16 bf16, KxN) in the contiguous K order used here:
//   lanes 0-15  : column n = nt*16 + lane,    K = kt*32 + 0..15  (consecutive)
//   lanes 16-31 : column n = nt*16 + lane-16, K = kt*32 + 16..31
// Each lane stores its 16 bf16 (32 bytes) contiguously:
//   frag[(kt*NTILES + nt)*32 + lane]
// ---------------------------------------------------------------------------
__global__ __launch_bounds__(256) void pack_wenc_bf16(
    const float* __restrict__ W, bf16x16* __restrict__ out)
{
    const int gid  = blockIdx.x * blockDim.x + threadIdx.x; // 0..16383
    const int tile = gid >> 5;            // 0..511 = kt*16 + nt
    const int lane = gid & 31;
    const int kt   = tile >> 4;
    const int nt   = tile & 15;
    const int n    = nt * 16 + (lane & 15);
    const int k0   = kt * KSTEP + ((lane >> 4) << 4);
    bf16x16 frag;
#pragma unroll
    for (int r = 0; r < 16; ++r) {
        frag[r] = (__bf16)W[(k0 + r) * Z_DIM + n];
    }
    out[tile * 32 + lane] = frag;
}

// ---------------------------------------------------------------------------
// Kernel 2: fused encoder GEMM (bf16 WMMA, f32 accum) + ReLU + attention score.
//
// Grid: ROWS_TOTAL/128 blocks of 256 threads (8 waves, 2 M-rows x 4 N-cols).
// Each wave owns a 64x64 tile: 4x4 accumulator tiles (128 VGPRs).
// Per k-step, the block double-buffer-stages into LDS:
//   - A as bf16 fragments (8 tiles x 32 lanes x 32B = 8 KB): pipelined
//     global f32 loads -> v_cvt_pk_bf16_f32 -> ds_store (done once, reused 4x)
//   - B fragments (16 KB) via global_load_async_to_lds_b128 (ASYNCcnt)
// A fragment (16x32 bf16) layout used here (contiguous K, as in B):
//   lanes 0-15  : row m = lane,    elems 0..15 = K 0..15
//   lanes 16-31 : row m = lane-16, elems 0..15 = K 16..31
// Epilogue: emb = relu(acc + b_enc); row score = sum_n emb*w_att reduced
// across lanes (shuffles) and across N-col waves (LDS ds_add_f32).
// ---------------------------------------------------------------------------
struct ARegs { float4 a0, a1, a2, a3; };

__device__ __forceinline__ ARegs load_a_regs(const float* rowp, int kt, int hi) {
    const float4* p0 = (const float4*)(rowp + kt * KSTEP + hi * 16);
    const float4* p1 = (const float4*)(rowp + kt * KSTEP + 8 + hi * 16);
    ARegs r;
    r.a0 = p0[0]; r.a1 = p0[1];
    r.a2 = p1[0]; r.a3 = p1[1];
    return r;
}

__device__ __forceinline__ void cvt_store_a(const ARegs& r, char* sAbuf, int slot) {
    bf16x16 f;
    f[0]  = (__bf16)r.a0.x; f[1]  = (__bf16)r.a0.y;
    f[2]  = (__bf16)r.a0.z; f[3]  = (__bf16)r.a0.w;
    f[4]  = (__bf16)r.a1.x; f[5]  = (__bf16)r.a1.y;
    f[6]  = (__bf16)r.a1.z; f[7]  = (__bf16)r.a1.w;
    f[8]  = (__bf16)r.a2.x; f[9]  = (__bf16)r.a2.y;
    f[10] = (__bf16)r.a2.z; f[11] = (__bf16)r.a2.w;
    f[12] = (__bf16)r.a3.x; f[13] = (__bf16)r.a3.y;
    f[14] = (__bf16)r.a3.z; f[15] = (__bf16)r.a3.w;
    ((bf16x16*)sAbuf)[slot] = f;
}

__device__ __forceinline__ void issue_b_async(const char* bsrc_kt, char* sBbuf, int tid) {
#pragma unroll
    for (int c = 0; c < 4; ++c) {
        const int chunk = tid * 4 + c;     // 0..1023 chunks of 16B = 16 KB
        async_copy_b128(bsrc_kt + chunk * 16, sBbuf + chunk * 16);
    }
}

__global__ __launch_bounds__(256) void encoder_score_kernel(
    const float*   __restrict__ bags,     // [ROWS_TOTAL, F]
    const bf16x16* __restrict__ wfrag,    // packed W_enc bf16 fragments
    const float*   __restrict__ b_enc,    // [Z]
    const float*   __restrict__ w_att,    // [Z]
    const float*   __restrict__ b_att,    // [1]
    float*         __restrict__ scores)   // [ROWS_TOTAL]
{
    __shared__ char  sA[2][A_KSTEP_BYTES];   // bf16 A fragments, double buffered
    __shared__ char  sB[2][B_KSTEP_BYTES];   // bf16 B fragments, double buffered
    __shared__ float s_score[ROWS_PER_BLOCK];

    const int tid  = threadIdx.x;
    const int lane = tid & 31;
    const int wave = tid >> 5;
    const int hi   = lane >> 4;     // K-half select
    const int nl   = lane & 15;     // row (A) / column (B,C) within tile
    const int waveM = wave >> 2;    // 0..1  -> M offset waveM*64
    const int waveN = wave & 3;     // 0..3  -> N offset waveN*64
    const int rowBase = blockIdx.x * ROWS_PER_BLOCK;

    if (tid < ROWS_PER_BLOCK) s_score[tid] = 0.0f;

    // This thread stages A fragment slot (tile = tid>>5, lane): one 16-row tile,
    // this lane's (m, K-half) slice of 16 f32 -> bf16.
    const float* myArow =
        bags + (size_t)(rowBase + (tid >> 5) * 16 + nl) * F_DIM;
    const char* bsrc = (const char*)wfrag;

    // ---- prologue: stage k-step 0 -----------------------------------------
    ARegs areg = load_a_regs(myArow, 0, hi);
    issue_b_async(bsrc, sB[0], tid);
    cvt_store_a(areg, sA[0], tid);
    wait_async_zero();
    __syncthreads();

    f32x8 acc[4][4];
#pragma unroll
    for (int mt = 0; mt < 4; ++mt)
#pragma unroll
        for (int nt = 0; nt < 4; ++nt)
#pragma unroll
            for (int j = 0; j < 8; ++j) acc[mt][nt][j] = 0.0f;

    for (int kt = 0; kt < KTILES; ++kt) {
        const int  cur  = kt & 1;
        const int  nxt  = cur ^ 1;
        const bool more = (kt + 1) < KTILES;

        // ---- kick off staging of the next k-step (overlaps with compute) --
        if (more) {
            areg = load_a_regs(myArow, kt + 1, hi);
            issue_b_async(bsrc + (size_t)(kt + 1) * B_KSTEP_BYTES, sB[nxt], tid);
        }

        // ---- compute current k-step: 4 A frags, 4 B frags, 16 WMMAs -------
        // Load all 8 fragments into distinct registers first so the
        // scheduler can overlap LDS latency with WMMA issue (partial
        // s_wait_dscnt instead of full drains between quartets).
        bf16x16 af[4], bfr[4];
#pragma unroll
        for (int mt = 0; mt < 4; ++mt)
            af[mt] = ((const bf16x16*)sA[cur])[(waveM * 4 + mt) * 32 + lane];
#pragma unroll
        for (int nt = 0; nt < 4; ++nt)
            bfr[nt] = ((const bf16x16*)sB[cur])[(waveN * 4 + nt) * 32 + lane];

#pragma unroll
        for (int nt = 0; nt < 4; ++nt) {
#pragma unroll
            for (int mt = 0; mt < 4; ++mt) {
                acc[mt][nt] = __builtin_amdgcn_wmma_f32_16x16x32_bf16(
                    false, af[mt], false, bfr[nt], (short)0, acc[mt][nt],
                    false, false);
            }
        }

        // ---- finish staging next buffers, then rotate ---------------------
        if (more) {
            cvt_store_a(areg, sA[nxt], tid);
            wait_async_zero();
        }
        __syncthreads();
    }

    // ---- Epilogue: bias + ReLU + dot with w_att ---------------------------
    // C layout: lanes 0-15: N=nl, VGPR r -> M=r ; lanes 16-31: N=nl, M=8+r
    const float batt = b_att[0];
#pragma unroll
    for (int mt = 0; mt < 4; ++mt) {
        float partial[8];
#pragma unroll
        for (int r = 0; r < 8; ++r) partial[r] = 0.0f;
#pragma unroll
        for (int nt = 0; nt < 4; ++nt) {
            const int   n  = waveN * 64 + nt * 16 + nl;
            const float wa = w_att[n];
            const float be = b_enc[n];
#pragma unroll
            for (int r = 0; r < 8; ++r) {
                float e = fmaxf(acc[mt][nt][r] + be, 0.0f);
                partial[r] = fmaf(e, wa, partial[r]);
            }
        }
#pragma unroll
        for (int r = 0; r < 8; ++r) {
            float v = partial[r];
            v += __shfl_xor(v, 1);
            v += __shfl_xor(v, 2);
            v += __shfl_xor(v, 4);
            v += __shfl_xor(v, 8);     // sum over 16 lanes (halves disjoint)
            if (nl == 0) {
                atomicAdd(&s_score[waveM * 64 + mt * 16 + hi * 8 + r], v);
            }
        }
    }
    __syncthreads();
    if (tid < ROWS_PER_BLOCK) {
        scores[rowBase + tid] = s_score[tid] + batt;
    }
}

// ---------------------------------------------------------------------------
// Kernel 3: per-batch top-K, softmax, recompute 20 embedding rows, pool,
// BatchNorm (eval) + head linear. One block of 256 threads per batch.
// ---------------------------------------------------------------------------
__global__ __launch_bounds__(256) void topk_pool_head_kernel(
    const float* __restrict__ scores,   // [B, N]
    const float* __restrict__ bags,     // [B, N, F]
    const float* __restrict__ W_enc,    // [F, Z]
    const float* __restrict__ b_enc,    // [Z]
    const float* __restrict__ bn_gamma,
    const float* __restrict__ bn_beta,
    const float* __restrict__ bn_mean,
    const float* __restrict__ bn_var,
    const float* __restrict__ W_head,   // [Z, 2]
    const float* __restrict__ b_head,   // [2]
    float*       __restrict__ out)      // [B, 2]
{
    const int b   = blockIdx.x;
    const int tid = threadIdx.x;

    __shared__ float s_val[256];
    __shared__ int   s_idx[256];
    __shared__ int   sel_idx[TOPK];
    __shared__ float sel_val[TOPK];
    __shared__ float attn[TOPK];
    __shared__ float s_chunk[TOPK][128];
    __shared__ float s_norm[Z_DIM];

    const float* sc = scores + (size_t)b * N_PATCH;

    // ---- iterative top-K argmax -------------------------------------------
    for (int t = 0; t < TOPK; ++t) {
        float lm = -INFINITY;
        int   li = -1;
        for (int i = tid; i < N_PATCH; i += 256) {
            bool taken = false;
            for (int j = 0; j < t; ++j) taken |= (sel_idx[j] == i);
            if (taken) continue;
            const float v = sc[i];
            if (v > lm || (v == lm && (unsigned)i < (unsigned)li)) { lm = v; li = i; }
        }
        s_val[tid] = lm; s_idx[tid] = li;
        __syncthreads();
        for (int s = 128; s > 0; s >>= 1) {
            if (tid < s) {
                const float ov = s_val[tid + s]; const int oi = s_idx[tid + s];
                const float mv = s_val[tid];     const int mi = s_idx[tid];
                if (ov > mv || (ov == mv && (unsigned)oi < (unsigned)mi)) {
                    s_val[tid] = ov; s_idx[tid] = oi;
                }
            }
            __syncthreads();
        }
        if (tid == 0) { sel_idx[t] = s_idx[0]; sel_val[t] = s_val[0]; }
        __syncthreads();
    }

    // ---- softmax over the K selected scores -------------------------------
    if (tid == 0) {
        float mx = sel_val[0];
        for (int t = 1; t < TOPK; ++t) mx = fmaxf(mx, sel_val[t]);
        float s = 0.0f;
        for (int t = 0; t < TOPK; ++t) {
            const float e = expf(sel_val[t] - mx);
            attn[t] = e; s += e;
        }
        const float inv = 1.0f / s;
        for (int t = 0; t < TOPK; ++t) attn[t] *= inv;
    }
    __syncthreads();

    // ---- recompute emb rows for the K winners; thread tid owns z = tid ----
    const int z = tid;   // Z_DIM == blockDim.x == 256
    float acc[TOPK];
#pragma unroll
    for (int t = 0; t < TOPK; ++t) acc[t] = 0.0f;

    for (int k0 = 0; k0 < F_DIM; k0 += 128) {
        for (int i = tid; i < TOPK * 128; i += 256) {
            const int t = i >> 7;
            const int k = i & 127;
            s_chunk[t][k] =
                bags[((size_t)b * N_PATCH + sel_idx[t]) * F_DIM + k0 + k];
        }
        __syncthreads();
        for (int k = 0; k < 128; ++k) {
            const float w = W_enc[(k0 + k) * Z_DIM + z];
#pragma unroll
            for (int t = 0; t < TOPK; ++t)
                acc[t] = fmaf(s_chunk[t][k], w, acc[t]);
        }
        __syncthreads();
    }

    const float be = b_enc[z];
    float pooled = 0.0f;
#pragma unroll
    for (int t = 0; t < TOPK; ++t) {
        const float e = fmaxf(acc[t] + be, 0.0f);
        pooled = fmaf(attn[t], e, pooled);
    }

    // ---- BatchNorm (eval) + head ------------------------------------------
    const float normed =
        (pooled - bn_mean[z]) * rsqrtf(bn_var[z] + BN_EPS) * bn_gamma[z] + bn_beta[z];
    s_norm[z] = normed;
    __syncthreads();

    if (tid < 2) {
        float o = b_head[tid];
        for (int zz = 0; zz < Z_DIM; ++zz)
            o = fmaf(s_norm[zz], W_head[zz * 2 + tid], o);
        out[b * 2 + tid] = o;
    }
}

// ---------------------------------------------------------------------------
// Launch
// ---------------------------------------------------------------------------
extern "C" void kernel_launch(void* const* d_in, const int* in_sizes, int n_in,
                              void* d_out, int out_size, void* d_ws, size_t ws_size,
                              hipStream_t stream) {
    const float* bags   = (const float*)d_in[0];
    const float* W_enc  = (const float*)d_in[1];
    const float* b_enc  = (const float*)d_in[2];
    const float* w_att  = (const float*)d_in[3];
    const float* b_att  = (const float*)d_in[4];
    const float* bn_g   = (const float*)d_in[5];
    const float* bn_b   = (const float*)d_in[6];
    const float* bn_m   = (const float*)d_in[7];
    const float* bn_v   = (const float*)d_in[8];
    const float* W_head = (const float*)d_in[9];
    const float* b_head = (const float*)d_in[10];
    float* out = (float*)d_out;

    bf16x16* wfrag  = (bf16x16*)d_ws;
    float*   scores = (float*)((char*)d_ws + WENC_FRAG_BYTES);

    // 1) repack W_enc to bf16 fragments (512 KB, stays hot in L2)
    pack_wenc_bf16<<<64, 256, 0, stream>>>(W_enc, wfrag);

    // 2) fused encoder GEMM + attention scores (the 68.7 GFLOP / 512 MB part)
    encoder_score_kernel<<<ROWS_TOTAL / ROWS_PER_BLOCK, 256, 0, stream>>>(
        bags, wfrag, b_enc, w_att, b_att, scores);

    // 3) per-batch top-K + softmax + pooled recompute + BN + head
    topk_pool_head_kernel<<<B_BATCH, 256, 0, stream>>>(
        scores, bags, W_enc, b_enc, bn_g, bn_b, bn_m, bn_v, W_head, b_head, out);
}
